// Block_7584912244953
// MI455X (gfx1250) — compile-verified, matrix operand order checked by reading, not run on hardware
//
#include <hip/hip_runtime.h>
#include <math.h>

// ---------------------------------------------------------------------------
// Types for CDNA5 WMMA (wave32)
// ---------------------------------------------------------------------------
typedef __attribute__((ext_vector_type(16))) __bf16 v16bf;
typedef __attribute__((ext_vector_type(4)))  __bf16 bf16x4;
typedef __attribute__((ext_vector_type(8)))  float  v8f;

// ---------------------------------------------------------------------------
// Generic GEMM: C(f32, MrxNc) = A(f32, MrxK, lda) * B(f32, KxNc, ldb)
//  - A tiles: DMA'd fp32 -> LDS via global_load_async_to_lds_b128 (ASYNCcnt),
//    converted to bf16 at fragment-load time (cvt co-executes with XDL).
//  - B tiles: register-staged (loads issued a tile ahead), converted to bf16,
//    stored transposed so fragment loads are contiguous ds_load_b128.
//  - Double-buffered LDS, one barrier per K step, 8 WMMAs per wave per step
//    (wave tile 32x64: A fragments reused 4x to amortize the cvt chain).
// Requires Mr%128==0, Nc%128==0, K%32==0 (true for every GEMM here).
// epilogue: 0 = none, 1 = bias + exact GELU, 2 = bias only
// ---------------------------------------------------------------------------
#define TM 128
#define TN 128
#define TK 32
#define APAD 4   // fp32 pad  -> row stride 36 floats (144B, 16B aligned)
#define BPAD 8   // bf16 pad  -> row stride 40 bf16  (80B, 16B aligned)

__global__ __launch_bounds__(256) void wmma_gemm_kernel(
    const float* __restrict__ A, const float* __restrict__ B,
    float* __restrict__ C,
    int Mr, int Nc, int Kd, int lda, int ldb, int ldc,
    const float* __restrict__ bias, int epilogue)
{
    __shared__ float  As[2][TM][TK + APAD];   // fp32 A tiles (async DMA target)
    __shared__ __bf16 Bs[2][TN][TK + BPAD];   // bf16 B tiles, transposed [n][k]

    const int tid  = threadIdx.x;
    const int wave = tid >> 5;
    const int lane = tid & 31;
    const int wm   = wave & 3;       // wave M sub-tile: wm*32
    const int wn   = wave >> 2;      // wave N sub-tile: wn*64
    const int half = lane >> 4;      // lane-half selects K chunk (ISA layout)
    const int l16  = lane & 15;

    const int rowBase = blockIdx.y * TM;
    const int colBase = blockIdx.x * TN;

    const int c0 = 8 * half;         // K chunks: lanes 0-15 -> {0..7,16..23}
    const int c1 = 16 + 8 * half;    //           lanes 16-31 -> {8..15,24..31}

    v8f acc[2][4] = {};              // [mt][nt] 16x16 f32 accumulators

    // ---- stage helpers -----------------------------------------------------
    // A: 128x32 fp32 tile, 4 async b128 per thread, lane-scattered LDS dest
    auto stageA = [&](int k0, int nb) {
        #pragma unroll
        for (int p = 0; p < 4; ++p) {
            int flat = (p * 256 + tid) * 4;            // 0..4095
            int r = flat >> 5;
            int c = flat & 31;
            const float* gp = A + (size_t)(rowBase + r) * lda + k0 + c;
            unsigned int lp = (unsigned int)(uintptr_t)&As[nb][r][c];
            asm volatile("global_load_async_to_lds_b128 %0, %1, off"
                         :: "v"(lp), "v"(gp) : "memory");
        }
    };
    // B: issue 16 scalar loads for the 32x128 tile slice owned by this thread
    auto loadB = [&](int k0, float* br) {
        #pragma unroll
        for (int p = 0; p < 4; ++p) {
            int idx = p * 256 + tid;                   // 0..1023
            int n  = idx & (TN - 1);                   // 0..127
            int kg = idx >> 7;                         // k = 4*kg .. +3
            const float* bp = B + (size_t)(k0 + kg * 4) * ldb + colBase + n;
            br[p * 4 + 0] = bp[0];
            br[p * 4 + 1] = bp[(size_t)ldb];
            br[p * 4 + 2] = bp[2 * (size_t)ldb];
            br[p * 4 + 3] = bp[3 * (size_t)ldb];
        }
    };
    auto storeB = [&](int nb, const float* br) {
        #pragma unroll
        for (int p = 0; p < 4; ++p) {
            int idx = p * 256 + tid;
            int n  = idx & (TN - 1);
            int kg = idx >> 7;
            bf16x4 bv = { (__bf16)br[p * 4 + 0], (__bf16)br[p * 4 + 1],
                          (__bf16)br[p * 4 + 2], (__bf16)br[p * 4 + 3] };
            *(bf16x4*)&Bs[nb][n][kg * 4] = bv;
        }
    };
    // A fragment: 16 fp32 from LDS -> v16bf (cvt co-executes with WMMA)
    auto afragLoad = [&](int bf, int r) -> v16bf {
        const float4 f0 = *(const float4*)&As[bf][r][c0];
        const float4 f1 = *(const float4*)&As[bf][r][c0 + 4];
        const float4 f2 = *(const float4*)&As[bf][r][c1];
        const float4 f3 = *(const float4*)&As[bf][r][c1 + 4];
        v16bf a;
        a[0]  = (__bf16)f0.x; a[1]  = (__bf16)f0.y; a[2]  = (__bf16)f0.z; a[3]  = (__bf16)f0.w;
        a[4]  = (__bf16)f1.x; a[5]  = (__bf16)f1.y; a[6]  = (__bf16)f1.z; a[7]  = (__bf16)f1.w;
        a[8]  = (__bf16)f2.x; a[9]  = (__bf16)f2.y; a[10] = (__bf16)f2.z; a[11] = (__bf16)f2.w;
        a[12] = (__bf16)f3.x; a[13] = (__bf16)f3.y; a[14] = (__bf16)f3.z; a[15] = (__bf16)f3.w;
        return a;
    };
    auto bfragLoad = [&](int bf, int n) -> v16bf {
        typedef __attribute__((ext_vector_type(8))) __bf16 v8bf;
        v8bf lo = *(const v8bf*)&Bs[bf][n][c0];
        v8bf hi = *(const v8bf*)&Bs[bf][n][c1];
        return __builtin_shufflevector(lo, hi, 0,1,2,3,4,5,6,7,8,9,10,11,12,13,14,15);
    };

    // ---- software pipeline -------------------------------------------------
    float breg[16];
    stageA(0, 0);
    loadB(0, breg);
    storeB(0, breg);
    asm volatile("s_wait_asynccnt 0x0" ::: "memory");
    __syncthreads();

    int buf = 0;
    for (int k0 = 0; k0 < Kd; k0 += TK) {
        const int  nbuf = buf ^ 1;
        const bool nxt  = (k0 + TK) < Kd;
        if (nxt) {                       // stage next tile while computing
            stageA(k0 + TK, nbuf);       // async DMA, no VGPR traffic
            loadB(k0 + TK, breg);        // loads in flight across the WMMAs
        }

        v16bf afrag[2], bfrag[4];
        #pragma unroll
        for (int mt = 0; mt < 2; ++mt)
            afrag[mt] = afragLoad(buf, wm * 32 + mt * 16 + l16);
        #pragma unroll
        for (int nt = 0; nt < 4; ++nt)
            bfrag[nt] = bfragLoad(buf, wn * 64 + nt * 16 + l16);

        #pragma unroll
        for (int mt = 0; mt < 2; ++mt)
            #pragma unroll
            for (int nt = 0; nt < 4; ++nt)
                acc[mt][nt] = __builtin_amdgcn_wmma_f32_16x16x32_bf16(
                    false, afrag[mt], false, bfrag[nt],
                    (short)0, acc[mt][nt], false, false);

        if (nxt) {
            storeB(nbuf, breg);          // single loadcnt wait lands here,
            asm volatile("s_wait_asynccnt 0x0" ::: "memory"); // behind WMMAs
        }
        __syncthreads();
        buf = nbuf;
    }

    // ---- epilogue + store (D layout: VGPR r -> M = r + 8*half, N = l16) ----
    #pragma unroll
    for (int mt = 0; mt < 2; ++mt) {
        #pragma unroll
        for (int nt = 0; nt < 4; ++nt) {
            int gn  = colBase + wn * 64 + nt * 16 + l16;
            int gm0 = rowBase + wm * 32 + mt * 16 + 8 * half;
            float bv = (epilogue != 0) ? bias[gn] : 0.0f;
            #pragma unroll
            for (int r = 0; r < 8; ++r) {
                float v = acc[mt][nt][r] + bv;
                if (epilogue == 1)
                    v = 0.5f * v * (1.0f + erff(v * 0.70710678118654752f)); // exact GELU
                C[(size_t)(gm0 + r) * ldc + gn] = v;
            }
        }
    }
}

// ---------------------------------------------------------------------------
// RMSNorm: one 256-thread block per row of 512
// ---------------------------------------------------------------------------
__global__ __launch_bounds__(256) void rmsnorm_kernel(
    const float* __restrict__ x, const float* __restrict__ g, float* __restrict__ o)
{
    const int row = blockIdx.x;
    const int tid = threadIdx.x;
    const float* xr = x + (size_t)row * 512;
    float a = xr[tid];
    float b = xr[tid + 256];
    float ss = a * a + b * b;
    #pragma unroll
    for (int m = 16; m >= 1; m >>= 1) ss += __shfl_xor(ss, m, 32);
    __shared__ float red[8];
    if ((tid & 31) == 0) red[tid >> 5] = ss;
    __syncthreads();
    float tot = 0.0f;
    #pragma unroll
    for (int i = 0; i < 8; ++i) tot += red[i];
    float r = rsqrtf(tot * (1.0f / 512.0f) + 1e-6f);
    o[(size_t)row * 512 + tid]       = a * r * g[tid];
    o[(size_t)row * 512 + tid + 256] = b * r * g[tid + 256];
}

// ---------------------------------------------------------------------------
// Fused per-node SDPA over 4 branches (H=8, D=64, 8x8 softmax) + sum +
// RMSNorm(g2). One block per node; wave h handles head h; lane covers
// d = {lane, lane+32}. st* = per-branch row stride for q/k (ee branch reads
// the packed 1024-wide projection buffer); v rows are always 512-wide.
// ---------------------------------------------------------------------------
__global__ __launch_bounds__(256) void sdpa_rms_kernel(
    const float* __restrict__ q0, const float* __restrict__ k0, const float* __restrict__ v0,
    const float* __restrict__ q1, const float* __restrict__ k1, const float* __restrict__ v1,
    const float* __restrict__ q2, const float* __restrict__ k2, const float* __restrict__ v2,
    const float* __restrict__ q3, const float* __restrict__ k3, const float* __restrict__ v3,
    int st0, int st1, int st2, int st3,
    const float* __restrict__ g2, float* __restrict__ y)
{
    const int n    = blockIdx.x;
    const int h    = threadIdx.x >> 5;
    const int lane = threadIdx.x & 31;
    const size_t rowOff = (size_t)n * 512;

    const float* Q[4] = { q0, q1, q2, q3 };
    const float* K[4] = { k0, k1, k2, k3 };
    const float* V[4] = { v0, v1, v2, v3 };
    const int   ST[4] = { st0, st1, st2, st3 };

    float acc0 = 0.0f, acc1 = 0.0f;
    const float scale = 0.125f;   // 1/sqrt(D=64)

    for (int b = 0; b < 4; ++b) {
        const size_t qkOff = (size_t)n * ST[b];
        const float* qp = Q[b] + qkOff + h * 64;
        float qa = qp[lane], qb = qp[lane + 32];
        float s[8];
        #pragma unroll
        for (int g = 0; g < 8; ++g) {
            const float* kp = K[b] + qkOff + g * 64;
            float p = qa * kp[lane] + qb * kp[lane + 32];
            #pragma unroll
            for (int m = 16; m >= 1; m >>= 1) p += __shfl_xor(p, m, 32);
            s[g] = p * scale;
        }
        float mx = s[0];
        #pragma unroll
        for (int g = 1; g < 8; ++g) mx = fmaxf(mx, s[g]);
        float sum = 0.0f;
        #pragma unroll
        for (int g = 0; g < 8; ++g) { s[g] = expf(s[g] - mx); sum += s[g]; }
        float inv = 1.0f / sum;
        #pragma unroll
        for (int g = 0; g < 8; ++g) {
            const float* vp = V[b] + rowOff + g * 64;
            float a = s[g] * inv;
            acc0 += a * vp[lane];
            acc1 += a * vp[lane + 32];
        }
    }

    float ss = acc0 * acc0 + acc1 * acc1;
    #pragma unroll
    for (int m = 16; m >= 1; m >>= 1) ss += __shfl_xor(ss, m, 32);
    __shared__ float red[8];
    if (lane == 0) red[h] = ss;
    __syncthreads();
    float tot = 0.0f;
    #pragma unroll
    for (int i = 0; i < 8; ++i) tot += red[i];
    float r = rsqrtf(tot * (1.0f / 512.0f) + 1e-6f);

    int cA = h * 64 + lane;
    int cB = cA + 32;
    y[rowOff + cA] = acc0 * r * g2[cA];
    y[rowOff + cB] = acc1 * r * g2[cB];
}

// ---------------------------------------------------------------------------
// Host-side orchestration
// ---------------------------------------------------------------------------
static inline void launch_gemm(const float* A, const float* B, float* C,
                               int Mr, int Nc, int Kd, int lda, int ldb, int ldc,
                               const float* bias, int ep, hipStream_t s)
{
    dim3 grid(Nc / TN, Mr / TM);
    wmma_gemm_kernel<<<grid, 256, 0, s>>>(A, B, C, Mr, Nc, Kd, lda, ldb, ldc, bias, ep);
}

extern "C" void kernel_launch(void* const* d_in, const int* in_sizes, int n_in,
                              void* d_out, int out_size, void* d_ws, size_t ws_size,
                              hipStream_t stream)
{
    const float* x_node = (const float*)d_in[0];
    const float* x_edge = (const float*)d_in[1];
    const float* adj    = (const float*)d_in[2];
    const float* Wq_hh  = (const float*)d_in[3];
    const float* Wk_hh  = (const float*)d_in[4];
    const float* Wv_hh  = (const float*)d_in[5];
    const float* Wq_ee  = (const float*)d_in[6];
    const float* Wk_ee  = (const float*)d_in[7];
    const float* Wv_ee  = (const float*)d_in[8];
    const float* Wq_he  = (const float*)d_in[9];
    const float* Wk_he  = (const float*)d_in[10];
    const float* Wv_he  = (const float*)d_in[11];
    const float* Wq_eh  = (const float*)d_in[12];
    const float* Wk_eh  = (const float*)d_in[13];
    const float* Wv_eh  = (const float*)d_in[14];
    const float* Wp1_he = (const float*)d_in[15];
    const float* Wp1_eh = (const float*)d_in[16];
    const float* Wp1_ee = (const float*)d_in[17];
    const float* g_n    = (const float*)d_in[18];
    const float* g_e    = (const float*)d_in[19];
    const float* g2     = (const float*)d_in[20];
    const float* Wf1    = (const float*)d_in[21];
    const float* bf1    = (const float*)d_in[22];
    const float* Wf2    = (const float*)d_in[23];
    const float* bf2    = (const float*)d_in[24];
    float* out = (float*)d_out;

    const size_t NE = (size_t)4096 * 512;    // node activation
    const size_t ME = (size_t)16384 * 512;   // edge activation (== 4096*2048)

    float* ws    = (float*)d_ws;
    float* h     = ws;            // (N,512)   rms(x_node)
    float* e     = h + NE;        // (M,512)   rms(x_edge)
    float* eqk_e = e + ME;        // (M,1024)  packed [e@Wq_ee | e@Wk_ee]
    float* eq_eh = eqk_e + 2*ME;  // (M,512)   e @ Wq_eh
    float* ek_he = eq_eh + ME;    // (M,512)   e @ Wk_he
    float* qpre  = ek_he + ME;    // (N,512)   h @ Wq_hh (pre-adjacency)
    float* k_hh  = qpre + NE;
    float* v_hh  = k_hh + NE;
    float* v_ee  = v_hh + NE;
    float* k_eh  = v_ee + NE;
    float* v_eh  = k_eh + NE;
    float* q_he  = v_eh + NE;
    float* v_he  = q_he + NE;
    float* q_hh  = v_he + NE;     // (N,512)   adj @ qpre
    float* qk_ee = q_hh + NE;     // (N,1024)  Wp1_ee @ eqk_e  (q | k packed)
    float* q_eh  = qk_ee + 2*NE;  // (N,512)   Wp1_eh @ eq_eh
    float* k_he  = q_eh + NE;     // (N,512)   Wp1_he @ ek_he
    float* y     = k_he + NE;     // (N,512)   rms(branch sum, g2)
    float* ffn_h = eqk_e;         // alias: edge buffers dead after projections

    // 1) RMSNorms
    rmsnorm_kernel<<<4096,  256, 0, stream>>>(x_node, g_n, h);
    rmsnorm_kernel<<<16384, 256, 0, stream>>>(x_edge, g_e, e);

    // 2) node-side QKV GEMMs: (4096,512) x (512,512)
    launch_gemm(h, Wq_hh, qpre, 4096, 512, 512, 512, 512, 512, nullptr, 0, stream);
    launch_gemm(h, Wk_hh, k_hh, 4096, 512, 512, 512, 512, 512, nullptr, 0, stream);
    launch_gemm(h, Wv_hh, v_hh, 4096, 512, 512, 512, 512, 512, nullptr, 0, stream);
    launch_gemm(h, Wv_ee, v_ee, 4096, 512, 512, 512, 512, 512, nullptr, 0, stream);
    launch_gemm(h, Wk_eh, k_eh, 4096, 512, 512, 512, 512, 512, nullptr, 0, stream);
    launch_gemm(h, Wv_eh, v_eh, 4096, 512, 512, 512, 512, 512, nullptr, 0, stream);
    launch_gemm(h, Wq_he, q_he, 4096, 512, 512, 512, 512, 512, nullptr, 0, stream);
    launch_gemm(h, Wv_he, v_he, 4096, 512, 512, 512, 512, 512, nullptr, 0, stream);

    // 3) edge-side QK GEMMs: (16384,512) x (512,512)
    //    ee pair written packed side-by-side (ldc=1024) so Wp1_ee is read once
    launch_gemm(e, Wq_ee, eqk_e,       16384, 512, 512, 512, 512, 1024, nullptr, 0, stream);
    launch_gemm(e, Wk_ee, eqk_e + 512, 16384, 512, 512, 512, 512, 1024, nullptr, 0, stream);
    launch_gemm(e, Wq_eh, eq_eh,       16384, 512, 512, 512, 512, 512,  nullptr, 0, stream);
    launch_gemm(e, Wk_he, ek_he,       16384, 512, 512, 512, 512, 512,  nullptr, 0, stream);

    // 4) shape projections (HBM-bound: 832 MB of Wp1/adj reads total)
    launch_gemm(adj,    qpre,  q_hh,  4096, 512,  4096,  4096,  512,  512,  nullptr, 0, stream);
    launch_gemm(Wp1_ee, eqk_e, qk_ee, 4096, 1024, 16384, 16384, 1024, 1024, nullptr, 0, stream);
    launch_gemm(Wp1_eh, eq_eh, q_eh,  4096, 512,  16384, 16384, 512,  512,  nullptr, 0, stream);
    launch_gemm(Wp1_he, ek_he, k_he,  4096, 512,  16384, 16384, 512,  512,  nullptr, 0, stream);

    // 5) fused 4-branch per-node SDPA + sum + RMSNorm(g2)
    sdpa_rms_kernel<<<4096, 256, 0, stream>>>(q_hh,  k_hh,        v_hh,
                                              qk_ee, qk_ee + 512, v_ee,
                                              q_eh,  k_eh,        v_eh,
                                              q_he,  k_he,        v_he,
                                              512, 1024, 512, 512,
                                              g2, y);

    // 6) FFN with fused bias+GELU / bias epilogues
    launch_gemm(y,     Wf1, ffn_h, 4096, 2048, 512,  512,  2048, 2048, bf1, 1, stream);
    launch_gemm(ffn_h, Wf2, out,   4096, 512,  2048, 2048, 512,  512,  bf2, 2, stream);
}
